// TemporalSelfAttention_54494545051890
// MI455X (gfx1250) — matrix operand order repaired and checked by reading, compiled, not run on hardware
//
#include <hip/hip_runtime.h>

// ---------------- types / constants ----------------
typedef __bf16 v16bf __attribute__((ext_vector_type(16)));
typedef float  v8f   __attribute__((ext_vector_type(8)));

#define Hh   8
#define Vn   25
#define Nn   16
#define Tt   512
#define Dd   1600            // D_IN == D_OUT
#define NH   (Nn*Hh)         // 128
#define NHTE ((size_t)NH*Tt*Dd)   // elements in one of Q/K/V
static const float ATT_SCALE = 0.025f;   // 1/sqrt(1600)

__device__ __forceinline__ __bf16 f2bf(float f) {
  unsigned u = __builtin_bit_cast(unsigned, f);
  u += 0x7FFFu + ((u >> 16) & 1u);            // round-to-nearest-even
  unsigned short h = (unsigned short)(u >> 16);
  return __builtin_bit_cast(__bf16, h);
}

__device__ __forceinline__ v8f wmma_bf16(v16bf a, v16bf b, v8f c) {
  return __builtin_amdgcn_wmma_f32_16x16x32_bf16(false, a, false, b, (short)0, c, false, false);
}

// ---------------- CDNA5 async global->LDS (16B per lane) ----------------
__device__ __forceinline__ void async_cp16(const __bf16* lds_dst, const __bf16* gsrc) {
  unsigned laddr = (unsigned)(size_t)(const void*)lds_dst;
  unsigned long long gaddr = (unsigned long long)(size_t)(const void*)gsrc;
  asm volatile("global_load_async_to_lds_b128 %0, %1, off"
               :: "v"(laddr), "v"(gaddr) : "memory");
}
template <int N>
__device__ __forceinline__ void wait_async() {
  asm volatile("s_wait_asynccnt %0" :: "i"(N) : "memory");
}

// 64 rows x 32 bf16 cols tile -> LDS stride 40 (pad 8), 2 chunks/thread
__device__ __forceinline__ void async_tile32(__bf16* dst, const __bf16* src,
                                             size_t sstride, int tid) {
#pragma unroll
  for (int i = 0; i < 2; ++i) {
    int c = tid + i * 128;
    int r = c >> 2, q = c & 3;
    async_cp16(dst + r * 40 + q * 8, src + (size_t)r * sstride + q * 8);
  }
}
// 32 rows x 64 bf16 cols tile -> LDS stride 72 (pad 8), 2 chunks/thread
__device__ __forceinline__ void async_tile64(__bf16* dst, const __bf16* src,
                                             size_t sstride, int tid) {
#pragma unroll
  for (int i = 0; i < 2; ++i) {
    int c = tid + i * 128;
    int r = c >> 3, q = c & 7;
    async_cp16(dst + r * 72 + q * 8, src + (size_t)r * sstride + q * 8);
  }
}

// A fragment: 16x32 bf16, lane l row m=l&15; elems 0..15 -> K=16*(l>>4)+j
__device__ __forceinline__ v16bf frag_a(const __bf16* As, int ldr, int wrow, int lane) {
  const __bf16* p = As + (wrow + (lane & 15)) * ldr;
  int ko = (lane >> 4) << 4;
  v16bf a;
#pragma unroll
  for (int j = 0; j < 8; ++j) { a[j] = p[ko + j]; a[j + 8] = p[ko + 8 + j]; }
  return a;
}
// B fragment from LDS tile stored [n][k] (k contiguous)
__device__ __forceinline__ v16bf frag_b_nk(const __bf16* Bs, int ldr, int ncol, int lane) {
  const __bf16* p = Bs + (ncol + (lane & 15)) * ldr + ((lane >> 4) << 4);
  v16bf b;
#pragma unroll
  for (int j = 0; j < 16; ++j) b[j] = p[j];
  return b;
}
// B fragment from LDS tile stored [k][n]
__device__ __forceinline__ v16bf frag_b_kn(const __bf16* Bs, int ldr, int ncol, int lane) {
  int n = ncol + (lane & 15);
  int ko = (lane >> 4) << 4;
  v16bf b;
#pragma unroll
  for (int j = 0; j < 16; ++j) b[j] = Bs[(ko + j) * ldr + n];
  return b;
}

// ---------------- kernel 1: pack x -> xf bf16 [n][t][c*25+v] ----------------
__global__ __launch_bounds__(256) void pack_x(const float* __restrict__ x,
                                              __bf16* __restrict__ xf) {
  int gid = blockIdx.x * 256 + threadIdx.x;
  if (gid >= Nn * Tt * Dd) return;
  int d = gid % Dd;
  int t = (gid / Dd) % Tt;
  int n = gid / (Dd * Tt);
  int c = d / Vn, v = d - c * Vn;
  xf[gid] = f2bf(x[(((size_t)n * 64 + c) * Tt + t) * Vn + v]);
}

// ---------------- kernel 2: W[h][d][e] f32 -> WT[w][h][e][d] bf16 ----------------
__global__ __launch_bounds__(256) void pack_w(const float* __restrict__ Wq,
                                              const float* __restrict__ Wk,
                                              const float* __restrict__ Wv,
                                              __bf16* __restrict__ WT) {
  __shared__ float tile[32][33];
  int wz = blockIdx.z;               // 0..23 : w*8+h
  int w3 = wz >> 3;
  const float* W = (w3 == 0 ? Wq : w3 == 1 ? Wk : Wv) + (size_t)(wz & 7) * Dd * Dd;
  int e0 = blockIdx.x * 32, d0 = blockIdx.y * 32;
  for (int r = threadIdx.y; r < 32; r += 8)
    tile[r][threadIdx.x] = W[(size_t)(d0 + r) * Dd + e0 + threadIdx.x];
  __syncthreads();
  __bf16* Wo = WT + (size_t)wz * Dd * Dd;
  for (int r = threadIdx.y; r < 32; r += 8)
    Wo[(size_t)(e0 + r) * Dd + d0 + threadIdx.x] = f2bf(tile[threadIdx.x][r]);
}

// ---------------- kernel 3: QKV projections (WMMA, async double-buffer) ----------------
__global__ __launch_bounds__(128) void qkv_gemm(const __bf16* __restrict__ xf,
                                                const __bf16* __restrict__ WT,
                                                const float* __restrict__ bq,
                                                const float* __restrict__ bk,
                                                const float* __restrict__ bv,
                                                __bf16* __restrict__ QKV) {
  __shared__ __align__(16) __bf16 As[2][64 * 40];
  __shared__ __align__(16) __bf16 Bs[2][64 * 40];
  int tile = blockIdx.x;                 // 0..199
  int tn = tile % 25, tm = tile / 25;
  int nh = blockIdx.y, h = nh & 7;
  int w3 = blockIdx.z;
  const float* bias = (w3 == 0) ? bq : (w3 == 1) ? bk : bv;
  const __bf16* Ab = xf + ((size_t)(nh >> 3) * Tt + tm * 64) * Dd;
  const __bf16* Wb = WT + ((size_t)w3 * 8 + h) * Dd * Dd + (size_t)tn * 64 * Dd;
  __bf16* Db = QKV + (size_t)w3 * NHTE + (size_t)nh * Tt * Dd;

  int tid = threadIdx.x, lane = tid & 31, wv = tid >> 5;
  v8f z = {0.f, 0.f, 0.f, 0.f, 0.f, 0.f, 0.f, 0.f};
  v8f acc[4] = {z, z, z, z};

  const int NK = Dd / 32;                      // 50
  async_tile32(As[0], Ab, Dd, tid);
  async_tile32(Bs[0], Wb, Dd, tid);
  for (int ki = 0; ki < NK; ++ki) {
    int cur = ki & 1, nxt = cur ^ 1;
    if (ki + 1 < NK) {
      async_tile32(As[nxt], Ab + (size_t)(ki + 1) * 32, Dd, tid);
      async_tile32(Bs[nxt], Wb + (size_t)(ki + 1) * 32, Dd, tid);
      wait_async<4>();                         // cur buffer complete
    } else {
      wait_async<0>();
    }
    __syncthreads();
    v16bf a = frag_a(As[cur], 40, wv * 16, lane);
#pragma unroll
    for (int f = 0; f < 4; ++f)
      acc[f] = wmma_bf16(a, frag_b_nk(Bs[cur], 40, f * 16, lane), acc[f]);
    __syncthreads();
  }
  int nloc = lane & 15, hi = lane >> 4;
#pragma unroll
  for (int f = 0; f < 4; ++f) {
    int e = tn * 64 + f * 16 + nloc;
    float bb = bias[h * Dd + e];
#pragma unroll
    for (int r = 0; r < 8; ++r) {
      int t = tm * 64 + wv * 16 + r + 8 * hi;
      Db[(size_t)t * Dd + e] = f2bf(acc[f][r] + bb);
    }
  }
}

// ---------------- kernel 4: S = scale * Q K^T  (WMMA, async double-buffer) ----------------
__global__ __launch_bounds__(128) void score_gemm(const __bf16* __restrict__ Q,
                                                  const __bf16* __restrict__ Km,
                                                  float* __restrict__ S) {
  __shared__ __align__(16) __bf16 As[2][64 * 40];
  __shared__ __align__(16) __bf16 Bs[2][64 * 40];  // K rows [s][e] -> B as [n=s][k=e]
  int tile = blockIdx.x;           // 0..63 : ts(8) * tt(8)
  int ts = tile & 7, tt = tile >> 3;
  int nh = blockIdx.y;
  const __bf16* Ab = Q  + ((size_t)nh * Tt + tt * 64) * Dd;
  const __bf16* Kb = Km + ((size_t)nh * Tt + ts * 64) * Dd;
  int tid = threadIdx.x, lane = tid & 31, wv = tid >> 5;
  v8f z = {0.f, 0.f, 0.f, 0.f, 0.f, 0.f, 0.f, 0.f};
  v8f acc[4] = {z, z, z, z};

  const int NK = Dd / 32;
  async_tile32(As[0], Ab, Dd, tid);
  async_tile32(Bs[0], Kb, Dd, tid);
  for (int ki = 0; ki < NK; ++ki) {
    int cur = ki & 1, nxt = cur ^ 1;
    if (ki + 1 < NK) {
      async_tile32(As[nxt], Ab + (size_t)(ki + 1) * 32, Dd, tid);
      async_tile32(Bs[nxt], Kb + (size_t)(ki + 1) * 32, Dd, tid);
      wait_async<4>();
    } else {
      wait_async<0>();
    }
    __syncthreads();
    v16bf a = frag_a(As[cur], 40, wv * 16, lane);
#pragma unroll
    for (int f = 0; f < 4; ++f)
      acc[f] = wmma_bf16(a, frag_b_nk(Bs[cur], 40, f * 16, lane), acc[f]);
    __syncthreads();
  }
  float* So = S + (size_t)nh * Tt * Tt;
  int nloc = lane & 15, hi = lane >> 4;
#pragma unroll
  for (int f = 0; f < 4; ++f) {
    int s = ts * 64 + f * 16 + nloc;
#pragma unroll
    for (int r = 0; r < 8; ++r) {
      int t = tt * 64 + wv * 16 + r + 8 * hi;
      So[(size_t)t * Tt + s] = acc[f][r] * ATT_SCALE;
    }
  }
}

// ---------------- kernel 5: row softmax (one wave32 per 512-row) ----------------
__global__ __launch_bounds__(256) void softmax_k(float* __restrict__ S) {
  int row  = (blockIdx.x * 256 + threadIdx.x) >> 5;   // 65536 rows
  int lane = threadIdx.x & 31;
  float* p = S + (size_t)row * Tt;
  float v[16];
  float m = -3.0e38f;
#pragma unroll
  for (int i = 0; i < 16; ++i) { v[i] = p[lane + i * 32]; m = fmaxf(m, v[i]); }
#pragma unroll
  for (int off = 16; off; off >>= 1) m = fmaxf(m, __shfl_xor(m, off));
  float s = 0.f;
#pragma unroll
  for (int i = 0; i < 16; ++i) { v[i] = __expf(v[i] - m); s += v[i]; }
#pragma unroll
  for (int off = 16; off; off >>= 1) s += __shfl_xor(s, off);
  float inv = 1.f / s;
#pragma unroll
  for (int i = 0; i < 16; ++i) p[lane + i * 32] = v[i] * inv;
}

// ---------------- kernel 6: O = P * V + permuted store ----------------
__global__ __launch_bounds__(128) void out_gemm(const float* __restrict__ S,
                                                const __bf16* __restrict__ Vm,
                                                float* __restrict__ out) {
  __shared__ __align__(16) __bf16 As[2][64 * 40];  // P tile, f32->bf16 via VALU
  __shared__ __align__(16) __bf16 Vs[2][32 * 72];  // V tile [k=s][n=e], async
  int tile = blockIdx.x;           // 0..199 : te(25) * tt(8)
  int te = tile % 25, tt = tile / 25;
  int nh = blockIdx.y, n = nh >> 3, h = nh & 7;
  const float*  Sb = S  + ((size_t)nh * Tt + tt * 64) * Tt;
  const __bf16* Vb = Vm + (size_t)nh * Tt * Dd + te * 64;
  int tid = threadIdx.x, lane = tid & 31, wv = tid >> 5;
  v8f z = {0.f, 0.f, 0.f, 0.f, 0.f, 0.f, 0.f, 0.f};
  v8f acc[4] = {z, z, z, z};

  const int NK = Tt / 32;                      // 16
  // prologue fill of buffer 0
  async_tile64(Vs[0], Vb, Dd, tid);
  for (int c = tid; c < 64 * 32; c += 128) {
    int r = c >> 5, col = c & 31;
    As[0][r * 40 + col] = f2bf(Sb[(size_t)r * Tt + col]);
  }
  for (int ki = 0; ki < NK; ++ki) {
    int cur = ki & 1, nxt = cur ^ 1;
    if (ki + 1 < NK) {
      int s0 = (ki + 1) * 32;
      async_tile64(Vs[nxt], Vb + (size_t)s0 * Dd, Dd, tid);
      for (int c = tid; c < 64 * 32; c += 128) {
        int r = c >> 5, col = c & 31;
        As[nxt][r * 40 + col] = f2bf(Sb[(size_t)r * Tt + s0 + col]);
      }
      wait_async<2>();
    } else {
      wait_async<0>();
    }
    __syncthreads();
    v16bf a = frag_a(As[cur], 40, wv * 16, lane);
#pragma unroll
    for (int f = 0; f < 4; ++f)
      acc[f] = wmma_bf16(a, frag_b_kn(Vs[cur], 72, f * 16, lane), acc[f]);
    __syncthreads();
  }
  // out[n][c*8+h][t][v]  where e = v*64 + c
  int nloc = lane & 15, hi = lane >> 4;
#pragma unroll
  for (int f = 0; f < 4; ++f) {
    int e = te * 64 + f * 16 + nloc;
    int vnode = e >> 6, c = e & 63;
    float* ob = out + ((size_t)n * 512 + (c * 8 + h)) * (size_t)Tt * Vn;
#pragma unroll
    for (int r = 0; r < 8; ++r) {
      int t = tt * 64 + wv * 16 + r + 8 * hi;
      ob[(size_t)t * Vn + vnode] = acc[f][r];
    }
  }
}

// ---------------- launcher ----------------
extern "C" void kernel_launch(void* const* d_in, const int* in_sizes, int n_in,
                              void* d_out, int out_size, void* d_ws, size_t ws_size,
                              hipStream_t stream) {
  const float* x  = (const float*)d_in[0];
  const float* Wq = (const float*)d_in[1];
  const float* bq = (const float*)d_in[2];
  const float* Wk = (const float*)d_in[3];
  const float* bk = (const float*)d_in[4];
  const float* Wv = (const float*)d_in[5];
  const float* bv = (const float*)d_in[6];
  float* out = (float*)d_out;

  char* ws = (char*)d_ws;
  const size_t XF_B  = (size_t)Nn * Tt * Dd * 2;           //  26,214,400
  const size_t WT_B  = (size_t)3 * 8 * Dd * Dd * 2;        // 122,880,000
  const size_t QKV_B = (size_t)3 * NHTE * 2;               // 629,145,600
  __bf16* xf  = (__bf16*)ws;
  __bf16* WT  = (__bf16*)(ws + XF_B);
  __bf16* QKV = (__bf16*)(ws + XF_B + WT_B);
  float*  S   = (float*)(ws + XF_B + WT_B + QKV_B);        // 134,217,728 -> ~912 MB total

  pack_x   <<<dim3((Nn * Tt * Dd + 255) / 256), dim3(256), 0, stream>>>(x, xf);
  pack_w   <<<dim3(50, 50, 24), dim3(32, 8), 0, stream>>>(Wq, Wk, Wv, WT);
  qkv_gemm <<<dim3(200, NH, 3), dim3(128), 0, stream>>>(xf, WT, bq, bk, bv, QKV);
  score_gemm<<<dim3(64, NH), dim3(128), 0, stream>>>(QKV, QKV + NHTE, S);
  softmax_k<<<dim3((NH * Tt) / 8), dim3(256), 0, stream>>>(S);
  out_gemm <<<dim3(200, NH), dim3(128), 0, stream>>>(S, QKV + 2 * NHTE, out);
}
